// CrossAttention_74672301408418
// MI455X (gfx1250) — matrix operand order repaired and checked by reading, not compiled, not run on hardware
//
#include <hip/hip_runtime.h>
#include <hip/hip_bf16.h>

// ---------------------------------------------------------------------------
// CDNA5 (gfx1250) spiking cross-attention, f16 WMMA pipeline.
// D = A(16x32 f16) x B(32x16 f16) + C(16x16 f32) via v_wmma_f32_16x16x32_f16.
// Attention K/V tiles staged in LDS with async copies (ASYNCcnt), ping-pong.
// ---------------------------------------------------------------------------

typedef __attribute__((ext_vector_type(16))) _Float16 v16h;
typedef __attribute__((ext_vector_type(8)))  _Float16 v8h;
typedef __attribute__((ext_vector_type(8)))  float    v8f;

#define TT 4
#define BB 2
#define NN 1024
#define CC 512
#define HH 8
#define DD 64

__device__ __forceinline__ v8f zero8() {
  v8f z;
#pragma unroll
  for (int i = 0; i < 8; ++i) z[i] = 0.0f;
  return z;
}

__device__ __forceinline__ v8f wmma16(v16h a, v16h b, v8f c) {
  // emits v_wmma_f32_16x16x32_f16
  return __builtin_amdgcn_wmma_f32_16x16x32_f16(
      /*neg_a=*/false, a, /*neg_b=*/false, b,
      /*c_mod=*/(short)0, c, /*reuse_a=*/false, /*reuse_b=*/false);
}

// Async 16B copy global -> LDS (GLOBAL_LOAD_ASYNC_TO_LDS_B128, ASYNCcnt).
__device__ __forceinline__ void async_copy_b128(void* lds_p, const void* g_p) {
  unsigned l = (unsigned)(uintptr_t)lds_p;  // low 32 bits = LDS byte address
  asm volatile("global_load_async_to_lds_b128 %0, %1, off"
               :: "v"(l), "v"(g_p) : "memory");
}

__device__ __forceinline__ void wait_asynccnt0() {
  asm volatile("s_wait_asynccnt 0x0" ::: "memory");
}

// A-matrix fragment (16x32 f16, M=lane%16): ISA layout:
//   halves[0..7]  = K = c0 + g*8  + (0..7)
//   halves[8..15] = K = c0 + 16 + g*8 + (0..7)     (g = lane/16)
__device__ __forceinline__ v16h load_a_frag(const _Float16* tile, int ld, int c0) {
  int l = threadIdx.x & 31;
  int m = l & 15, g = l >> 4;
  const _Float16* p = tile + m * ld + c0 + g * 8;
  v8h lo = *(const v8h*)(p);
  v8h hi = *(const v8h*)(p + 16);
  v16h r;
#pragma unroll
  for (int i = 0; i < 8; ++i) { r[i] = lo[i]; r[8 + i] = hi[i]; }
  return r;
}

// B-matrix fragment (32x16 f16, N=lane%16): B[k][n] = tile[n*ld + c0 + k].
// ISA layout: halves[0..7] = K = c0 + g*8 + (0..7), halves[8..15] = K = c0 + 16 + g*8 + (0..7).
__device__ __forceinline__ v16h load_b_frag(const _Float16* tile, int ld, int c0) {
  int l = threadIdx.x & 31;
  int n = l & 15, g = l >> 4;
  const _Float16* p = tile + n * ld + c0 + g * 8;
  v8h lo = *(const v8h*)(p);
  v8h hi = *(const v8h*)(p + 16);
  v16h r;
#pragma unroll
  for (int i = 0; i < 8; ++i) { r[i] = lo[i]; r[8 + i] = hi[i]; }
  return r;
}

// C/D layout: lane l, vgpr r -> row = (l>>4)*8 + r, col = l&15.

// ---------------------------------------------------------------------------
// Kernel 0: weights -> f16, BN params -> (scale, shift)
// ---------------------------------------------------------------------------
__global__ void __launch_bounds__(256)
prep_kernel(const float* __restrict__ Wq, const float* __restrict__ Wk,
            const float* __restrict__ Wv, const float* __restrict__ Wo,
            const float* __restrict__ bn,
            _Float16* __restrict__ W16, float* __restrict__ scaleA,
            float* __restrict__ shiftA) {
  int gid = blockIdx.x * 256 + threadIdx.x;
  if (gid < 4 * CC * CC) {
    int p = gid >> 18;            // 2^18 = 262144 = 512*512
    int idx = gid & (CC * CC - 1);
    const float* src = (p == 0) ? Wq : (p == 1) ? Wk : (p == 2) ? Wv : Wo;
    W16[gid] = (_Float16)src[idx];
  }
  if (gid < 4 * CC) {
    int p = gid >> 9, c = gid & (CC - 1);
    const float* b = bn + p * 4 * CC;
    float ga = b[c], be = b[CC + c], mn = b[2 * CC + c], vr = b[3 * CC + c];
    float sc = ga * rsqrtf(vr + 1e-5f);
    scaleA[gid] = sc;
    shiftA[gid] = be - mn * sc;
  }
}

// ---------------------------------------------------------------------------
// Kernel 1: Q/K/V projections.  grid = (512 row-tiles, 3 projections),
// 256 threads = 8 waves; wave w owns 64 output cols = head w.
// y = clamp(BN(x @ W^T), 0, 1) -> f16, head-split layouts.
// ---------------------------------------------------------------------------
__global__ void __launch_bounds__(256)
qkv_proj_kernel(const float* __restrict__ qin, const float* __restrict__ kin,
                const float* __restrict__ vin, const _Float16* __restrict__ W16,
                const float* __restrict__ scaleA, const float* __restrict__ shiftA,
                _Float16* __restrict__ q16, _Float16* __restrict__ k16,
                _Float16* __restrict__ v16T) {
  __shared__ __align__(16) _Float16 xt[16 * CC];

  int p = blockIdx.y;
  const float* x = (p == 0) ? qin : (p == 1) ? kin : vin;
  long rowBase = (long)blockIdx.x * 16;             // flat (t,b,n) row
  const float* xrow = x + rowBase * CC;

  // stage 16 x 512 f32 -> f16 in LDS (coalesced)
  for (int i = threadIdx.x; i < 16 * CC; i += 256)
    xt[i] = (_Float16)xrow[i];
  __syncthreads();

  int w = threadIdx.x >> 5;
  int l = threadIdx.x & 31, g = l >> 4, n16 = l & 15;
  const _Float16* Wp = W16 + (long)p * CC * CC + (long)(w * 64) * CC;

  v8f acc0 = zero8(), acc1 = zero8(), acc2 = zero8(), acc3 = zero8();
#pragma unroll 4
  for (int kc = 0; kc < CC; kc += 32) {
    v16h a = load_a_frag(xt, CC, kc);
    acc0 = wmma16(a, load_b_frag(Wp + 0 * 16 * CC, CC, kc), acc0);
    acc1 = wmma16(a, load_b_frag(Wp + 1 * 16 * CC, CC, kc), acc1);
    acc2 = wmma16(a, load_b_frag(Wp + 2 * 16 * CC, CC, kc), acc2);
    acc3 = wmma16(a, load_b_frag(Wp + 3 * 16 * CC, CC, kc), acc3);
  }

  int tb = (int)(rowBase >> 10);       // all 16 rows share (t,b)
  int nbase = (int)(rowBase & (NN - 1));
  const float* scp = scaleA + p * CC + w * 64;
  const float* shp = shiftA + p * CC + w * 64;

  v8f accs[4] = {acc0, acc1, acc2, acc3};
#pragma unroll
  for (int ct = 0; ct < 4; ++ct) {
    int dcol = ct * 16 + n16;               // 0..63 within head w
    float sc = scp[dcol], sh = shp[dcol];
#pragma unroll
    for (int r = 0; r < 8; ++r) {
      float y = accs[ct][r] * sc + sh;
      y = fminf(fmaxf(y, 0.0f), 1.0f);      // spike clamp
      int n = nbase + g * 8 + r;
      long tbh = (long)tb * HH + w;
      if (p == 0)      q16[(tbh * NN + n) * DD + dcol] = (_Float16)y;
      else if (p == 1) k16[(tbh * NN + n) * DD + dcol] = (_Float16)y;
      else             v16T[(tbh * DD + dcol) * NN + n] = (_Float16)y;   // transposed
    }
  }
}

// ---------------------------------------------------------------------------
// Kernel 2: flash attention per (t,b,h).  grid = (16 q-tiles, 64 tbh),
// 128 threads = 4 waves; wave owns 16 query rows, loops keys in blocks of 32.
// K/V tiles staged in LDS via async copies, double-buffered so the copy of
// tile i+1 overlaps the WMMA work on tile i.
// ---------------------------------------------------------------------------
__global__ void __launch_bounds__(128)
attn_kernel(const _Float16* __restrict__ q16, const _Float16* __restrict__ k16,
            const _Float16* __restrict__ v16T, _Float16* __restrict__ o16) {
  __shared__ __align__(16) _Float16 kt[2][32 * DD];     // 32 keys x 64 d
  __shared__ __align__(16) _Float16 vt[2][DD * 32];     // 64 d x 32 keys
  __shared__ __align__(16) _Float16 pbuf[4][16 * 32];   // per-wave P tile

  int tbh = blockIdx.y;
  int tid = threadIdx.x;
  int w = tid >> 5, l = tid & 31, g = l >> 4, n16 = l & 15;
  int nq0 = blockIdx.x * 64 + w * 16;

  const _Float16* qb = q16 + (long)tbh * NN * DD + (long)nq0 * DD;
  const _Float16* kb = k16 + (long)tbh * NN * DD;
  const _Float16* vb = v16T + (long)tbh * DD * NN;

  // stage one 32-key step: K tile is 4KB contiguous; V tile is 64 rows x 64B
  auto issue_copy = [&](int buf, int kbk) {
    const char* gk = (const char*)(kb + (long)kbk * DD);
    char* lk = (char*)&kt[buf][0];
    async_copy_b128(lk + tid * 16, gk + tid * 16);
    async_copy_b128(lk + (tid + 128) * 16, gk + (tid + 128) * 16);
    const char* gv = (const char*)vb + (long)kbk * 2;
    char* lv = (char*)&vt[buf][0];
    int c0 = tid, c1 = tid + 128;               // 256 chunks of 16B
    async_copy_b128(lv + (c0 >> 2) * 64 + (c0 & 3) * 16,
                    gv + (long)(c0 >> 2) * 2048 + (c0 & 3) * 16);
    async_copy_b128(lv + (c1 >> 2) * 64 + (c1 & 3) * 16,
                    gv + (long)(c1 >> 2) * 2048 + (c1 & 3) * 16);
  };

  v16h aq0 = load_a_frag(qb, DD, 0);
  v16h aq1 = load_a_frag(qb, DD, 32);

  v8f o0 = zero8(), o1 = zero8(), o2 = zero8(), o3 = zero8();
  float mrun[8], lrun[8], corr[8];
#pragma unroll
  for (int r = 0; r < 8; ++r) { mrun[r] = -1e30f; lrun[r] = 0.0f; }

  const float scale = 0.125f;  // D^-0.5, D = 64
  _Float16* pw = pbuf[w];

  issue_copy(0, 0);
  for (int it = 0; it < NN / 32; ++it) {
    int cur = it & 1;
    wait_asynccnt0();            // our async copies have landed in LDS
    __syncthreads();             // every wave's copies have landed
    if (it + 1 < NN / 32) issue_copy(1 - cur, (it + 1) * 32);  // overlap next

    const _Float16* ktc = &kt[cur][0];
    const _Float16* vtc = &vt[cur][0];

    // S tiles: rows = 16 queries, cols = 32 keys (two 16x16 tiles), K = D = 64
    v8f s0 = zero8(), s1 = zero8();
    s0 = wmma16(aq0, load_b_frag(ktc, DD, 0), s0);
    s0 = wmma16(aq1, load_b_frag(ktc, DD, 32), s0);
    s1 = wmma16(aq0, load_b_frag(ktc + 16 * DD, DD, 0), s1);
    s1 = wmma16(aq1, load_b_frag(ktc + 16 * DD, DD, 32), s1);

    // online softmax; row stats reduced across the 16 lanes of each half
#pragma unroll
    for (int r = 0; r < 8; ++r) {
      float x0 = s0[r] * scale, x1 = s1[r] * scale;
      float mx = fmaxf(x0, x1);
      mx = fmaxf(mx, __shfl_xor(mx, 1));
      mx = fmaxf(mx, __shfl_xor(mx, 2));
      mx = fmaxf(mx, __shfl_xor(mx, 4));
      mx = fmaxf(mx, __shfl_xor(mx, 8));
      float mnew = fmaxf(mrun[r], mx);
      float p0 = __expf(x0 - mnew), p1 = __expf(x1 - mnew);
      float rs = p0 + p1;
      rs += __shfl_xor(rs, 1);
      rs += __shfl_xor(rs, 2);
      rs += __shfl_xor(rs, 4);
      rs += __shfl_xor(rs, 8);
      float c = __expf(mrun[r] - mnew);
      lrun[r] = lrun[r] * c + rs;
      mrun[r] = mnew;
      corr[r] = c;
      // C-layout -> LDS (row = g*8+r, col = key offset)
      pw[(g * 8 + r) * 32 + n16]      = (_Float16)p0;
      pw[(g * 8 + r) * 32 + 16 + n16] = (_Float16)p1;
    }

    // DS ops are in-order per wave; fence the compiler + wait counter
    asm volatile("s_wait_dscnt 0x0" ::: "memory");
    v16h ap = load_a_frag(pw, 32, 0);   // P as A-fragment (16x32)

#pragma unroll
    for (int r = 0; r < 8; ++r) {
      o0[r] *= corr[r]; o1[r] *= corr[r]; o2[r] *= corr[r]; o3[r] *= corr[r];
    }
    // O += P @ V : B[k][n] = vt[(d0+n)*32 + k]
    o0 = wmma16(ap, load_b_frag(vtc + 0 * 16 * 32, 32, 0), o0);
    o1 = wmma16(ap, load_b_frag(vtc + 1 * 16 * 32, 32, 0), o1);
    o2 = wmma16(ap, load_b_frag(vtc + 2 * 16 * 32, 32, 0), o2);
    o3 = wmma16(ap, load_b_frag(vtc + 3 * 16 * 32, 32, 0), o3);

    __syncthreads();             // all waves done reading kt/vt[cur]
  }

  int tb = tbh >> 3, h = tbh & 7;
  _Float16* ob = o16 + (long)tb * NN * CC;
#pragma unroll
  for (int r = 0; r < 8; ++r) {
    float inv = 1.0f / lrun[r];
    int n = nq0 + g * 8 + r;
    long off = (long)n * CC + h * DD + n16;   // merge-heads layout [n][h*64+d]
    ob[off + 0]  = (_Float16)(o0[r] * inv);
    ob[off + 16] = (_Float16)(o1[r] * inv);
    ob[off + 32] = (_Float16)(o2[r] * inv);
    ob[off + 48] = (_Float16)(o3[r] * inv);
  }
}

// ---------------------------------------------------------------------------
// Kernel 3: output projection, BN (no clamp), f32 output.
// ---------------------------------------------------------------------------
__global__ void __launch_bounds__(256)
out_proj_kernel(const _Float16* __restrict__ o16, const _Float16* __restrict__ W16,
                const float* __restrict__ scaleA, const float* __restrict__ shiftA,
                float* __restrict__ out) {
  long rowBase = (long)blockIdx.x * 16;
  int w = threadIdx.x >> 5, l = threadIdx.x & 31, g = l >> 4, n16 = l & 15;

  const _Float16* ob = o16 + rowBase * CC;
  const _Float16* Wp = W16 + (long)3 * CC * CC + (long)(w * 64) * CC;

  v8f acc0 = zero8(), acc1 = zero8(), acc2 = zero8(), acc3 = zero8();
#pragma unroll 4
  for (int kc = 0; kc < CC; kc += 32) {
    v16h a = load_a_frag(ob, CC, kc);
    acc0 = wmma16(a, load_b_frag(Wp + 0 * 16 * CC, CC, kc), acc0);
    acc1 = wmma16(a, load_b_frag(Wp + 1 * 16 * CC, CC, kc), acc1);
    acc2 = wmma16(a, load_b_frag(Wp + 2 * 16 * CC, CC, kc), acc2);
    acc3 = wmma16(a, load_b_frag(Wp + 3 * 16 * CC, CC, kc), acc3);
  }

  const float* scp = scaleA + 3 * CC + w * 64;
  const float* shp = shiftA + 3 * CC + w * 64;
  v8f accs[4] = {acc0, acc1, acc2, acc3};
#pragma unroll
  for (int ct = 0; ct < 4; ++ct) {
    int c = ct * 16 + n16;
    float sc = scp[c], sh = shp[c];
#pragma unroll
    for (int r = 0; r < 8; ++r) {
      long n = rowBase + g * 8 + r;
      out[n * CC + w * 64 + c] = accs[ct][r] * sc + sh;
    }
  }
}

// ---------------------------------------------------------------------------
extern "C" void kernel_launch(void* const* d_in, const int* in_sizes, int n_in,
                              void* d_out, int out_size, void* d_ws, size_t ws_size,
                              hipStream_t stream) {
  (void)in_sizes; (void)n_in; (void)out_size; (void)ws_size;

  const float* q   = (const float*)d_in[0];
  const float* k   = (const float*)d_in[1];
  const float* v   = (const float*)d_in[2];
  const float* Wq  = (const float*)d_in[3];
  const float* Wk  = (const float*)d_in[4];
  const float* Wv  = (const float*)d_in[5];
  const float* Wo  = (const float*)d_in[6];
  const float* bn  = (const float*)d_in[7];
  float* out = (float*)d_out;

  char* ws = (char*)d_ws;
  const size_t OFF_W16   = 0;                       // 4*512*512*2   = 2 MiB
  const size_t OFF_SCALE = 2097152;                 // 2048*4
  const size_t OFF_SHIFT = 2097152 + 8192;
  const size_t OFF_Q16   = 2113536;                 // 8 MiB each
  const size_t OFF_K16   = OFF_Q16 + 8388608;
  const size_t OFF_VT    = OFF_K16 + 8388608;
  const size_t OFF_O16   = OFF_VT + 8388608;

  _Float16* W16    = (_Float16*)(ws + OFF_W16);
  float*    scaleA = (float*)(ws + OFF_SCALE);
  float*    shiftA = (float*)(ws + OFF_SHIFT);
  _Float16* q16    = (_Float16*)(ws + OFF_Q16);
  _Float16* k16    = (_Float16*)(ws + OFF_K16);
  _Float16* v16T   = (_Float16*)(ws + OFF_VT);
  _Float16* o16    = (_Float16*)(ws + OFF_O16);

  prep_kernel<<<4096, 256, 0, stream>>>(Wq, Wk, Wv, Wo, bn, W16, scaleA, shiftA);
  qkv_proj_kernel<<<dim3(512, 3), 256, 0, stream>>>(q, k, v, W16, scaleA, shiftA,
                                                    q16, k16, v16T);
  attn_kernel<<<dim3(16, 64), 128, 0, stream>>>(q16, k16, v16T, o16);
  out_proj_kernel<<<512, 256, 0, stream>>>(o16, W16, scaleA, shiftA, out);
}
